// MLA_8847632629906
// MI455X (gfx1250) — matrix operand order repaired and checked by reading, not compiled, not run on hardware
//
#include <hip/hip_runtime.h>
#include <hip/hip_bf16.h>

// ---------------------------------------------------------------------------
// MLA forward for MI455X (gfx1250, wave32, WMMA).
// All matmuls in bf16 WMMA (v_wmma_f32_16x16x32_bf16, f32 accumulate).
// GEMM B-tiles staged through LDS with async-to-LDS (ASYNCcnt); the software
// pipeline is unrolled x2 so all double-buffer indices are compile-time
// constants (avoids promote-alloca demoting fragments to LDS).
// ---------------------------------------------------------------------------

typedef __attribute__((ext_vector_type(16))) __bf16 v16bf;
typedef __attribute__((ext_vector_type(8)))  float  v8f;
typedef __attribute__((ext_vector_type(4)))  unsigned int u32x4;
typedef __attribute__((ext_vector_type(4)))  int v4i;

#if defined(__has_builtin)
#if __has_builtin(__builtin_amdgcn_global_load_async_to_lds_b128) && \
    __has_builtin(__builtin_amdgcn_s_wait_asynccnt)
#define ASYNC_LDS 1
#endif
#endif

union Frag {
    v16bf v;
    u32x4 q[2];
};

static __device__ __forceinline__ unsigned short f2bf(float f) {
    unsigned u = __float_as_uint(f);
    unsigned r = u + 0x7fffu + ((u >> 16) & 1u);   // round-to-nearest-even
    return (unsigned short)(r >> 16);
}
static __device__ __forceinline__ float bf2f(unsigned short h) {
    return __uint_as_float(((unsigned)h) << 16);
}

static __device__ __forceinline__ v8f wmma_bf16(v16bf a, v16bf b, v8f c) {
    // (neg_a, A, neg_b, B, c_mod, C, reuse_a, reuse_b)
    return __builtin_amdgcn_wmma_f32_16x16x32_bf16(false, a, false, b,
                                                   (short)0, c, false, false);
}

// ---------------------------------------------------------------------------
// Elementwise conversion kernels
// ---------------------------------------------------------------------------
__global__ void k_f32_to_bf16(const float* __restrict__ in,
                              unsigned short* __restrict__ out, int n) {
    int i = blockIdx.x * blockDim.x + threadIdx.x;
    if (i < n) out[i] = f2bf(in[i]);
}

// W[K,N] fp32 -> Wt[N,K] bf16 (transposed so WMMA B-fragments load contiguously)
__global__ void k_transpose_bf16(const float* __restrict__ W,
                                 unsigned short* __restrict__ Wt, int K, int N) {
    int i = blockIdx.x * blockDim.x + threadIdx.x;
    if (i < K * N) {
        int k = i / N;
        int n = i - k * N;
        Wt[(size_t)n * K + k] = f2bf(W[i]);
    }
}

// ---------------------------------------------------------------------------
// WMMA GEMM: C = A[M,K](bf16) * Bt[N,K]^T(bf16) + bias, epilogue->bf16.
// Block = 8 waves; each wave computes a 16x64 tile; block tile = 128x64.
// All 8 waves share one 64-col B tile -> stage it in LDS (async, double
// buffered) so global B traffic drops 8x and WMMAs overlap the copies.
// K is a multiple of 64 for every call site, so the pipeline is unrolled
// by two 32-wide k-steps per iteration with constant buffer indices.
// Output layout modes:
//   0: plain row-major [M,N]
//   1: q/k content interleave: col=h*128+d -> row*3072 + h*192 + d
//   2: q rope interleave:      col=h*64+dp -> row*3072 + h*192 + 128 + dp
//   3: V transposed:           col=h*128+d -> ((b*16+h)*128+d)*S + s
// ---------------------------------------------------------------------------
__global__ __launch_bounds__(256) void k_gemm_bf16(
    const unsigned short* __restrict__ A,
    const unsigned short* __restrict__ Bt,
    const float* __restrict__ bias,
    unsigned short* __restrict__ C,
    int M, int N, int K, int mode, int S_) {

    const int lane = threadIdx.x;
    const int wave = threadIdx.y;
    const int hi = lane >> 4;      // which 16-lane half
    const int ln = lane & 15;
    const int rowBase = blockIdx.y * 128 + wave * 16;
    const int colBase = blockIdx.x * 64;

    v8f acc[4] = {};

    // A fragment: row = ln, elems 0..15 = K(16*hi .. 16*hi+15)
    const unsigned short* aRow = A + (size_t)(rowBase + ln) * K + 16 * hi;

#ifdef ASYNC_LDS
    // ---- async-LDS staged B path -----------------------------------------
    __shared__ unsigned short bld[2][64][32];     // double-buffered 64x32 tile
    const int tid  = wave * 32 + lane;
    const int bcol = tid >> 2;                    // 0..63
    const int bko  = (tid & 3) * 8;               // 0,8,16,24 (x 2B = 16B)
    const unsigned short* bsrc = Bt + (size_t)(colBase + bcol) * K + bko;

    typedef __attribute__((address_space(1))) v4i gv4i;   // global int4*
    typedef __attribute__((address_space(3))) v4i lv4i;   // LDS int4*

    Frag af0, af1;

    // stage buffer `buf` with the 64x32 B tile at k-offset k
    auto stage = [&](int buf, int k) {
        __builtin_amdgcn_global_load_async_to_lds_b128(
            (gv4i*)(const void*)(bsrc + k),
            (lv4i*)(void*)&bld[buf][bcol][bko], 0, 0);
    };
    auto loadA = [&](Frag& f, int k) {
        f.q[0] = *(const u32x4*)(aRow + k);
        f.q[1] = *(const u32x4*)(aRow + k + 8);
    };
    auto compute = [&](const unsigned short (*bt)[32], const Frag& af) {
#pragma unroll
        for (int nt = 0; nt < 4; ++nt) {
            Frag bfr;   // 16 contiguous K values at 16*hi, from LDS
            bfr.q[0] = *(const u32x4*)(&bt[nt * 16 + ln][16 * hi]);
            bfr.q[1] = *(const u32x4*)(&bt[nt * 16 + ln][16 * hi + 8]);
            acc[nt] = wmma_bf16(af.v, bfr.v, acc[nt]);
        }
    };

    // prologue: fill buffer 0 for k=0
    stage(0, 0);
    loadA(af0, 0);
    __builtin_amdgcn_s_wait_asynccnt(0);
    __syncthreads();

    for (int k0 = 0; k0 < K; k0 += 64) {
        // phase 0: stage k0+32 into buffer 1 while computing on buffer 0
        stage(1, k0 + 32);
        loadA(af1, k0 + 32);
        __builtin_prefetch(aRow + k0 + 64, 0, 1);
        compute(bld[0], af0);
        __builtin_amdgcn_s_wait_asynccnt(0);
        __syncthreads();                  // buf1 ready; readers of buf0 done

        // phase 1: stage k0+64 into buffer 0 while computing on buffer 1
        const bool more = (k0 + 64) < K;
        if (more) {
            stage(0, k0 + 64);
            loadA(af0, k0 + 64);
        }
        compute(bld[1], af1);
        if (more) {
            __builtin_amdgcn_s_wait_asynccnt(0);
            __syncthreads();              // buf0 ready; readers of buf1 done
        }
    }
#else
    // ---- fallback: register double-buffered direct loads (x2 unrolled) ---
    Frag af0, af1, bf0[4], bf1[4];
    auto loadA = [&](Frag& f, int k) {
        f.q[0] = *(const u32x4*)(aRow + k);
        f.q[1] = *(const u32x4*)(aRow + k + 8);
    };
    auto loadB = [&](Frag* bf, int k) {
#pragma unroll
        for (int nt = 0; nt < 4; ++nt) {
            const unsigned short* bRow =
                Bt + (size_t)(colBase + nt * 16 + ln) * K + k + 16 * hi;
            bf[nt].q[0] = ((const u32x4*)bRow)[0];
            bf[nt].q[1] = ((const u32x4*)bRow)[1];
        }
    };
    auto compute = [&](const Frag& af, const Frag* bf) {
#pragma unroll
        for (int nt = 0; nt < 4; ++nt)
            acc[nt] = wmma_bf16(af.v, bf[nt].v, acc[nt]);
    };

    loadA(af0, 0);
    loadB(bf0, 0);
    for (int k0 = 0; k0 < K; k0 += 64) {
        loadA(af1, k0 + 32);
        loadB(bf1, k0 + 32);
        __builtin_prefetch(aRow + k0 + 64, 0, 1);
        compute(af0, bf0);
        const bool more = (k0 + 64) < K;
        if (more) {
            loadA(af0, k0 + 64);
            loadB(bf0, k0 + 64);
        }
        compute(af1, bf1);
    }
#endif

    // ---- epilogue: bias add, bf16 convert, layout-fused store ------------
#pragma unroll
    for (int nt = 0; nt < 4; ++nt) {
        int col = colBase + nt * 16 + ln;
        float bv = bias[col];
#pragma unroll
        for (int r = 0; r < 8; ++r) {
            int row = rowBase + r + 8 * hi;   // C layout: M = r + 8*hi
            unsigned short h = f2bf(acc[nt][r] + bv);
            size_t idx;
            if (mode == 0) {
                idx = (size_t)row * N + col;
            } else if (mode == 1) {
                idx = (size_t)row * 3072 + (size_t)(col >> 7) * 192 + (col & 127);
            } else if (mode == 2) {
                idx = (size_t)row * 3072 + (size_t)(col >> 6) * 192 + 128 + (col & 63);
            } else {
                int b = row / S_;
                int s = row - b * S_;
                idx = (((size_t)b * 16 + (col >> 7)) * 128 + (col & 127)) * (size_t)S_ + s;
            }
            C[idx] = h;
        }
    }
}

// ---------------------------------------------------------------------------
// RoPE on q_r slices, in place on qbuf [B*S, H, 192] (cols 128..191)
// ---------------------------------------------------------------------------
__global__ void k_rope_q(unsigned short* __restrict__ Qb, int S_, int n) {
    int idx = blockIdx.x * blockDim.x + threadIdx.x;
    if (idx >= n) return;
    int i = idx & 31;               // rotation index 0..31
    int h = (idx >> 5) & 15;        // head
    int token = idx >> 9;
    int pos = token % S_;
    float inv = __powf(10000.0f, -(float)i / 32.0f);
    float sn, cs;
    __sincosf((float)pos * inv, &sn, &cs);
    size_t base = (size_t)token * 3072 + (size_t)h * 192 + 128;
    float x1 = bf2f(Qb[base + i]);
    float x2 = bf2f(Qb[base + 32 + i]);
    Qb[base + i]      = f2bf(x1 * cs - x2 * sn);
    Qb[base + 32 + i] = f2bf(x1 * sn + x2 * cs);
}

// RoPE on shared k_r [B*S, 64], broadcast into kbuf head slots
__global__ void k_rope_k(const unsigned short* __restrict__ Kr,
                         unsigned short* __restrict__ Kb, int S_, int n) {
    int idx = blockIdx.x * blockDim.x + threadIdx.x;
    if (idx >= n) return;
    int i = idx & 31;
    int token = idx >> 5;
    int pos = token % S_;
    float inv = __powf(10000.0f, -(float)i / 32.0f);
    float sn, cs;
    __sincosf((float)pos * inv, &sn, &cs);
    float x1 = bf2f(Kr[(size_t)token * 64 + i]);
    float x2 = bf2f(Kr[(size_t)token * 64 + 32 + i]);
    unsigned short o1 = f2bf(x1 * cs - x2 * sn);
    unsigned short o2 = f2bf(x1 * sn + x2 * cs);
#pragma unroll
    for (int h = 0; h < 16; ++h) {
        size_t base = (size_t)token * 3072 + (size_t)h * 192 + 128;
        Kb[base + i]      = o1;
        Kb[base + 32 + i] = o2;
    }
}

// ---------------------------------------------------------------------------
// Flash attention (causal). 4 waves/block; wave owns 16 q rows.
// Q,K: [B*S, H, 192] bf16; Vt: [B, H, 128, S] bf16; Out fp32 [B,S,2048].
// Per 32-wide k chunk: 12 WMMA for Q*K^T, 8 WMMA for P*V.
// K/V fragments are register double-buffered (constant indices after full
// unroll) so loads for step dc+1 overlap the WMMA of step dc; next chunk's
// cachelines are prefetched.
// P is transposed S-layout -> A-layout via a per-wave LDS tile (wave-local,
// DScnt keeps same-wave LDS ops in order; no block barrier => no deadlock
// despite causal per-wave trip counts).
// ---------------------------------------------------------------------------
__global__ __launch_bounds__(128) void k_attn(
    const unsigned short* __restrict__ Q,
    const unsigned short* __restrict__ Kb,
    const unsigned short* __restrict__ Vt,
    float* __restrict__ Out, int S_) {

    __shared__ unsigned short plds[4][16][32];   // per-wave 16x32 bf16 P tile

    const int lane = threadIdx.x;
    const int wave = threadIdx.y;
    const int hi = lane >> 4;
    const int ln = lane & 15;
    const int h = blockIdx.y;
    const int b = blockIdx.z;
    const int qbase = blockIdx.x * 64 + wave * 16;
    const float scale = 0.07216878364870323f;    // 1/sqrt(192)

    // Q fragments for this wave's 16 rows, all 6 d-chunks (192 = 6*32)
    Frag qf[6];
    {
        const unsigned short* qrow =
            Q + ((size_t)(b * S_ + qbase + ln) * 16 + h) * 192 + 16 * hi;
#pragma unroll
        for (int dc = 0; dc < 6; ++dc) {
            qf[dc].q[0] = *(const u32x4*)(qrow + dc * 32);
            qf[dc].q[1] = *(const u32x4*)(qrow + dc * 32 + 8);
        }
    }

    v8f acc[8] = {};        // 16 q rows x 128 v dims
    float m[8], l[8];
#pragma unroll
    for (int r = 0; r < 8; ++r) { m[r] = -__builtin_inff(); l[r] = 0.0f; }

    const int kend = qbase + 16;                 // causal frontier (exclusive)
    for (int c = 0; c < kend; c += 32) {
        // ---- S tile = Q * K^T : two 16x16 accumulators over 32 k cols ----
        v8f sacc[2] = {};
#pragma unroll
        for (int t = 0; t < 2; ++t) {
            const unsigned short* krow =
                Kb + ((size_t)(b * S_ + c + t * 16 + ln) * 16 + h) * 192 + 16 * hi;
            if (c + 32 < kend)                    // next chunk, same lane row+32
                __builtin_prefetch(krow + (size_t)32 * 3072, 0, 1);
            Frag kf[2];
            kf[0].q[0] = ((const u32x4*)krow)[0];
            kf[0].q[1] = ((const u32x4*)krow)[1];
#pragma unroll
            for (int dc = 0; dc < 6; ++dc) {
                if (dc < 5) {   // double buffer: fetch dc+1 while computing dc
                    kf[(dc + 1) & 1].q[0] = ((const u32x4*)(krow + (dc + 1) * 32))[0];
                    kf[(dc + 1) & 1].q[1] = ((const u32x4*)(krow + (dc + 1) * 32))[1];
                }
                sacc[t] = wmma_bf16(qf[dc].v, kf[dc & 1].v, sacc[t]);
            }
        }

        // ---- online softmax (rows live across 16-lane halves) ----
        float p0[8], p1[8];
#pragma unroll
        for (int r = 0; r < 8; ++r) {
            int qpos = qbase + r + 8 * hi;
            float s0 = (c + ln      <= qpos) ? sacc[0][r] * scale : -__builtin_inff();
            float s1 = (c + 16 + ln <= qpos) ? sacc[1][r] * scale : -__builtin_inff();
            float rm = fmaxf(s0, s1);
            rm = fmaxf(rm, __shfl_xor(rm, 1));
            rm = fmaxf(rm, __shfl_xor(rm, 2));
            rm = fmaxf(rm, __shfl_xor(rm, 4));
            rm = fmaxf(rm, __shfl_xor(rm, 8));
            float mnew = fmaxf(m[r], rm);
            float alpha = __expf(m[r] - mnew);
            float e0 = __expf(s0 - mnew);        // exp(-inf)=0 handles masked
            float e1 = __expf(s1 - mnew);
            p0[r] = e0; p1[r] = e1;
            float rs = e0 + e1;
            rs += __shfl_xor(rs, 1);
            rs += __shfl_xor(rs, 2);
            rs += __shfl_xor(rs, 4);
            rs += __shfl_xor(rs, 8);
            l[r] = l[r] * alpha + rs;
            m[r] = mnew;
#pragma unroll
            for (int dt = 0; dt < 8; ++dt) acc[dt][r] *= alpha;
        }

        // ---- transpose P (C-layout) into A-layout through LDS ----
#pragma unroll
        for (int r = 0; r < 8; ++r) {
            plds[wave][r + 8 * hi][ln]      = f2bf(p0[r]);
            plds[wave][r + 8 * hi][16 + ln] = f2bf(p1[r]);
        }
        __builtin_amdgcn_wave_barrier();
        Frag pf;
        pf.q[0] = *(const u32x4*)(&plds[wave][ln][16 * hi]);
        pf.q[1] = *(const u32x4*)(&plds[wave][ln][16 * hi + 8]);
        __builtin_amdgcn_wave_barrier();

        // ---- O += P * V  (V^T rows are contiguous in s) ----
        const unsigned short* vbase =
            Vt + (((size_t)(b * 16 + h) * 128) + ln) * (size_t)S_ + c + 16 * hi;
        Frag vf[2];
        vf[0].q[0] = ((const u32x4*)vbase)[0];
        vf[0].q[1] = ((const u32x4*)vbase)[1];
#pragma unroll
        for (int dt = 0; dt < 8; ++dt) {
            if (dt < 7) {
                const unsigned short* vrow = vbase + (size_t)(dt + 1) * 16 * S_;
                vf[(dt + 1) & 1].q[0] = ((const u32x4*)vrow)[0];
                vf[(dt + 1) & 1].q[1] = ((const u32x4*)vrow)[1];
            }
            if (c + 32 < kend)
                __builtin_prefetch(vbase + (size_t)dt * 16 * S_ + 32, 0, 1);
            acc[dt] = wmma_bf16(pf.v, vf[dt & 1].v, acc[dt]);
        }
    }

    // ---- normalize and store fp32 ----
#pragma unroll
    for (int dt = 0; dt < 8; ++dt) {
        int d = dt * 16 + ln;
#pragma unroll
        for (int r = 0; r < 8; ++r) {
            int qpos = qbase + r + 8 * hi;
            Out[(size_t)(b * S_ + qpos) * 2048 + h * 128 + d] = acc[dt][r] / l[r];
        }
    }
}

// ---------------------------------------------------------------------------
// Host launcher
// ---------------------------------------------------------------------------
extern "C" void kernel_launch(void* const* d_in, const int* in_sizes, int n_in,
                              void* d_out, int out_size, void* d_ws, size_t ws_size,
                              hipStream_t stream) {
    (void)in_sizes; (void)n_in; (void)out_size; (void)ws_size;

    const float* x     = (const float*)d_in[0];
    const float* Wq_d  = (const float*)d_in[1];
    const float* bq_d  = (const float*)d_in[2];
    const float* Wq_uc = (const float*)d_in[3];
    const float* bq_uc = (const float*)d_in[4];
    const float* Wq_up = (const float*)d_in[5];
    const float* bq_up = (const float*)d_in[6];
    const float* Wkv_d = (const float*)d_in[7];
    const float* bkv_d = (const float*)d_in[8];
    const float* Wk_uc = (const float*)d_in[9];
    const float* bk_uc = (const float*)d_in[10];
    const float* Wk_up = (const float*)d_in[11];
    const float* bk_up = (const float*)d_in[12];
    const float* Wv_u  = (const float*)d_in[13];
    const float* bv_u  = (const float*)d_in[14];
    float* out = (float*)d_out;

    constexpr int B = 4, S = 2048, D = 2048, LQ = 1536, LKV = 512;
    constexpr int M = B * S;

    char* ws = (char*)d_ws;
    size_t off = 0;
    auto alloc = [&](size_t elems) -> unsigned short* {
        unsigned short* p = (unsigned short*)(ws + off);
        off += ((elems * 2 + 255) & ~(size_t)255);
        return p;
    };
    unsigned short* xb      = alloc((size_t)M * D);
    unsigned short* wq_d_t  = alloc((size_t)LQ * D);
    unsigned short* wq_uc_t = alloc((size_t)2048 * LQ);
    unsigned short* wq_up_t = alloc((size_t)1024 * LQ);
    unsigned short* wkv_d_t = alloc((size_t)LKV * D);
    unsigned short* wk_uc_t = alloc((size_t)2048 * LKV);
    unsigned short* wk_up_t = alloc((size_t)64 * LKV);
    unsigned short* wv_u_t  = alloc((size_t)2048 * LKV);
    unsigned short* qlat    = alloc((size_t)M * LQ);
    unsigned short* kvlat   = alloc((size_t)M * LKV);
    unsigned short* qbuf    = alloc((size_t)M * 3072);
    unsigned short* kbuf    = alloc((size_t)M * 3072);
    unsigned short* krtmp   = alloc((size_t)M * 64);
    unsigned short* vt      = alloc((size_t)M * 2048);

    const dim3 blk256(256);
    // 1) x -> bf16
    k_f32_to_bf16<<<(M * (D / 256)), blk256, 0, stream>>>(x, xb, M * D);
    // 2) weights -> transposed bf16
    k_transpose_bf16<<<(D * LQ) / 256, blk256, 0, stream>>>(Wq_d, wq_d_t, D, LQ);
    k_transpose_bf16<<<(LQ * 2048) / 256, blk256, 0, stream>>>(Wq_uc, wq_uc_t, LQ, 2048);
    k_transpose_bf16<<<(LQ * 1024) / 256, blk256, 0, stream>>>(Wq_up, wq_up_t, LQ, 1024);
    k_transpose_bf16<<<(D * LKV) / 256, blk256, 0, stream>>>(Wkv_d, wkv_d_t, D, LKV);
    k_transpose_bf16<<<(LKV * 2048) / 256, blk256, 0, stream>>>(Wk_uc, wk_uc_t, LKV, 2048);
    k_transpose_bf16<<<(LKV * 64) / 256, blk256, 0, stream>>>(Wk_up, wk_up_t, LKV, 64);
    k_transpose_bf16<<<(LKV * 2048) / 256, blk256, 0, stream>>>(Wv_u, wv_u_t, LKV, 2048);

    const dim3 gblk(32, 8);
    // 3) down projections
    k_gemm_bf16<<<dim3(LQ / 64, M / 128), gblk, 0, stream>>>(xb, wq_d_t, bq_d, qlat, M, LQ, D, 0, S);
    k_gemm_bf16<<<dim3(LKV / 64, M / 128), gblk, 0, stream>>>(xb, wkv_d_t, bkv_d, kvlat, M, LKV, D, 0, S);
    // 4) up projections (fused interleave / transpose epilogues)
    k_gemm_bf16<<<dim3(2048 / 64, M / 128), gblk, 0, stream>>>(qlat, wq_uc_t, bq_uc, qbuf, M, 2048, LQ, 1, S);
    k_gemm_bf16<<<dim3(1024 / 64, M / 128), gblk, 0, stream>>>(qlat, wq_up_t, bq_up, qbuf, M, 1024, LQ, 2, S);
    k_gemm_bf16<<<dim3(2048 / 64, M / 128), gblk, 0, stream>>>(kvlat, wk_uc_t, bk_uc, kbuf, M, 2048, LKV, 1, S);
    k_gemm_bf16<<<dim3(64 / 64, M / 128), gblk, 0, stream>>>(kvlat, wk_up_t, bk_up, krtmp, M, 64, LKV, 0, S);
    k_gemm_bf16<<<dim3(2048 / 64, M / 128), gblk, 0, stream>>>(kvlat, wv_u_t, bv_u, vt, M, 2048, LKV, 3, S);
    // 5) RoPE
    k_rope_q<<<(M * 16 * 32) / 256, blk256, 0, stream>>>(qbuf, S, M * 16 * 32);
    k_rope_k<<<(M * 32) / 256, blk256, 0, stream>>>(krtmp, kbuf, S, M * 32);
    // 6) causal flash attention
    k_attn<<<dim3(S / 64, 16, B), dim3(32, 4), 0, stream>>>(qbuf, kbuf, vt, out, S);
}
